// Encoder_66872640799015
// MI455X (gfx1250) — matrix operand order repaired and checked by reading, not compiled, not run on hardware
//
#include <hip/hip_runtime.h>
#include <hip/hip_bf16.h>

// CDNA5 / gfx1250 WMMA implementation of the recurrent trajectory encoder.
// One workgroup (128 threads = 4 wave32) owns TWO scenes (16 peds) and keeps
// the whole 12-step recurrence resident in LDS as bf16. Convs are im2col
// GEMMs on v_wmma_f32_16x16x32_bf16. Activations are quantized to bf16 once
// at production; A-fragments are assembled with 2x ds_load_b128 each (zero
// conversion VALU in the K-loop). B-fragments (weight-normed, pre-swizzled by
// a prep kernel into d_ws) are hoisted into registers for all 12 steps.

typedef __attribute__((ext_vector_type(16))) __bf16 v16bf;
typedef __attribute__((ext_vector_type(16))) short  v16s;
typedef __attribute__((ext_vector_type(8)))  float  v8f;
typedef __attribute__((ext_vector_type(8)))  int    v8i;
typedef __attribute__((ext_vector_type(4)))  int    v4i;

#define BATCH   32768
#define SEQLEN  12
#define NPED    16         // peds per workgroup == 2 scenes
// d_ws layout (ushort elements): gemm1 frags [0,12288), gemm2 [12288,18432), gemm3 [18432,21504)
#define WF2_OFF 12288
#define WF3_OFF 18432

// fp32 -> bf16 round-to-nearest-even (bit pattern as ushort)
__device__ __forceinline__ unsigned short f2bf(float f) {
  unsigned u = __float_as_uint(f);
  u += 0x7fffu + ((u >> 16) & 1u);
  return (unsigned short)(u >> 16);
}

// B-operand (32x16) K placement: lanes 0-15 hold K {0..7,16..23}, lanes 16-31 hold K {8..15,24..31}
__device__ __forceinline__ int kloc_b(int pair, int hk) {
  return (pair < 4) ? 8 * hk + 2 * pair : 16 + 8 * hk + 2 * (pair - 4);
}

__device__ __forceinline__ v16bf load_bfrag(const unsigned short* wf, int tile, int lane) {
  const v16s bs = *(const v16s*)(wf + ((size_t)tile * 32 + lane) * 16);
  return __builtin_bit_cast(v16bf, bs);
}

// Assemble a 16x32 bf16 A-fragment from bf16 LDS activations laid out
// [row][c] (c fastest, rowStride ushorts). K layout k = dk*C + c.
// Lanes 0-15 hold K {0..7,16..23}, lanes 16-31 hold K {8..15,24..31}:
// run0 = pairs 0..3 at k = kbase+0..7, run1 = pairs 4..7 at k = kbase+16..23,
// each run is 4 consecutive channel-pairs => one ds_load_b128.
template <int LOG2C>
__device__ __forceinline__ v16bf load_afrag(const unsigned short* actb, int rowBase,
                                            int rowStride, int kt, int hk) {
  const int CM = (1 << LOG2C) - 1;
  const int kb0 = kt * 32 + 8 * hk;
  const int kb1 = kb0 + 16;
  const int c0 = kb0 & CM, dk0 = kb0 >> LOG2C;
  const int c1 = kb1 & CM, dk1 = kb1 >> LOG2C;
  const v4i a0 = *(const v4i*)(actb + (rowBase + dk0) * rowStride + c0);
  const v4i a1 = *(const v4i*)(actb + (rowBase + dk1) * rowStride + c1);
  const v8i av = __builtin_shufflevector(a0, a1, 0, 1, 2, 3, 4, 5, 6, 7);
  return __builtin_bit_cast(v16bf, av);
}

// ---------------------------------------------------------------------------
// Prep kernel: weight-norm (w = g*v/||v||) + pre-swizzle into fragment order.
// wf[((kt*NT + nt)*32 + lane)*16 + i] holds B[k][o] as bf16, where
// k = kt*32 + kloc_b(i>>1,lane>>4) + (i&1), o = nt*16 + (lane&15).
// K index layout: k = dk*C + c.
// ---------------------------------------------------------------------------
__global__ void prep_weights(const float* __restrict__ v1, const float* __restrict__ g1,
                             const float* __restrict__ v2, const float* __restrict__ g2,
                             const float* __restrict__ v3, const float* __restrict__ g3,
                             unsigned short* __restrict__ wf) {
  __shared__ float n1[64], n2[32], n3[32];
  const int t = threadIdx.x;
  if (t < 64) { float s = 0.f; for (int j = 0; j < 192; ++j) { float x = v1[t*192 + j]; s += x*x; }
                n1[t] = g1[t] * rsqrtf(s); }
  if (t < 32) { float s = 0.f; for (int j = 0; j < 192; ++j) { float x = v2[t*192 + j]; s += x*x; }
                n2[t] = g2[t] * rsqrtf(s);
                float s3 = 0.f; for (int j = 0; j < 96; ++j) { float x = v3[t*96 + j]; s3 += x*x; }
                n3[t] = g3[t] * rsqrtf(s3); }
  __syncthreads();
  // GEMM1: 6 K-tiles x 4 N-tiles, C=64
  for (int idx = t; idx < 6*4*32*16; idx += blockDim.x) {
    int i = idx & 15, lane = (idx >> 4) & 31, nt = (idx >> 9) & 3, kt = idx >> 11;
    int n = lane & 15, hk = lane >> 4, o = nt*16 + n;
    int k = kt*32 + kloc_b(i >> 1, hk) + (i & 1);
    int c = k & 63, dk = k >> 6;
    wf[idx] = f2bf(v1[(o*64 + c)*3 + dk] * n1[o]);
  }
  // GEMM2: 6 K-tiles x 2 N-tiles, C=64
  for (int idx = t; idx < 6*2*32*16; idx += blockDim.x) {
    int i = idx & 15, lane = (idx >> 4) & 31, nt = (idx >> 9) & 1, kt = idx >> 10;
    int n = lane & 15, hk = lane >> 4, o = nt*16 + n;
    int k = kt*32 + kloc_b(i >> 1, hk) + (i & 1);
    int c = k & 63, dk = k >> 6;
    wf[WF2_OFF + idx] = f2bf(v2[(o*64 + c)*3 + dk] * n2[o]);
  }
  // GEMM3: 3 K-tiles x 2 N-tiles, C=32
  for (int idx = t; idx < 3*2*32*16; idx += blockDim.x) {
    int i = idx & 15, lane = (idx >> 4) & 31, nt = (idx >> 9) & 1, kt = idx >> 10;
    int n = lane & 15, hk = lane >> 4, o = nt*16 + n;
    int k = kt*32 + kloc_b(i >> 1, hk) + (i & 1);
    int c = k & 31, dk = k >> 5;
    wf[WF3_OFF + idx] = f2bf(v3[(o*32 + c)*3 + dk] * n3[o]);
  }
}

// ---------------------------------------------------------------------------
// Main kernel: one block = 2 scenes (16 peds). All state in LDS for 12 steps.
// ---------------------------------------------------------------------------
__global__ void __launch_bounds__(128)
encoder_kernel(const float* __restrict__ obs_traj,
               const float* __restrict__ W_se, const float* __restrict__ b_se,
               const float* __restrict__ b1,   const float* __restrict__ b2,
               const float* __restrict__ b3,
               const float* __restrict__ W_hp, const float* __restrict__ b_hp,
               const unsigned short* __restrict__ wf,
               float* __restrict__ out) {
  __shared__ __align__(16) unsigned short act[NPED*8*64];  // obs_emb bf16 [p][t=0..7][c]
  __shared__ __align__(16) unsigned short s1 [NPED*6*64];  // conv1 out bf16 [p][t=0..5][c]
  __shared__ __align__(16) unsigned short s2 [NPED*4*32];  // conv2 out bf16 [p][t=0..3][c]
  __shared__ float sb [NPED*64];     // s = conv3 reshaped fp32 [p][h], h = o*2 + t
  __shared__ float mx [2*64];        // per-scene segment max (2 scenes)
  __shared__ float relb[NPED*2];

  const int tid  = threadIdx.x;
  const int lane = tid & 31;
  const int wv   = tid >> 5;
  const int n    = lane & 15;        // fragment col (B/C/D) or row (A)
  const int hk   = lane >> 4;
  const int ped0 = blockIdx.x * NPED;

  // ---- hoist step-invariant B fragments into registers (15 x 8 VGPRs)
  const int nt1 = wv;                       // GEMM1: wave -> N-tile
  const int mt2 = wv >> 1, nt2 = wv & 1;    // GEMM2: wave -> (mt base, nt)
  const int mt3 = wv >> 1, nt3 = wv & 1;    // GEMM3: wave -> (mt, nt)
  v16bf b1f[6], b2f[6], b3f[3];
#pragma unroll
  for (int kt = 0; kt < 6; ++kt) b1f[kt] = load_bfrag(wf,           kt*4 + nt1, lane);
#pragma unroll
  for (int kt = 0; kt < 6; ++kt) b2f[kt] = load_bfrag(wf + WF2_OFF, kt*2 + nt2, lane);
#pragma unroll
  for (int kt = 0; kt < 3; ++kt) b3f[kt] = load_bfrag(wf + WF3_OFF, kt*2 + nt3, lane);
  const float bias1 = b1[nt1*16 + n];
  const float bias2 = b2[nt2*16 + n];
  const float bias3 = b3[nt3*16 + n];

  // ---- initial spatial embedding: obs_emb = bf16(obs_traj @ W_se^T + b_se)
  for (int idx = tid; idx < NPED*8*64; idx += 128) {
    int c = idx & 63, rem = idx >> 6, tt = rem & 7, p = rem >> 3;
    float x0 = obs_traj[((size_t)tt*BATCH + ped0 + p)*2 + 0];
    float x1 = obs_traj[((size_t)tt*BATCH + ped0 + p)*2 + 1];
    act[idx] = f2bf(x0*W_se[c*2] + x1*W_se[c*2 + 1] + b_se[c]);
  }
  __syncthreads();

  for (int step = 0; step < SEQLEN; ++step) {
    // ===== GEMM1: (96 x 192) x (192 x 64); wave = nt1, mt = 0..5
    {
      const int o = nt1*16 + n;
#pragma unroll
      for (int mt = 0; mt < 6; ++mt) {
        const int r = mt*16 + n;               // A row = (p, t)
        const int p = r / 6, t0 = r - p*6;
        v8f acc = {};
#pragma unroll
        for (int kt = 0; kt < 6; ++kt) {
          const v16bf a = load_afrag<6>(act, p*8 + t0, 64, kt, hk);
          acc = __builtin_amdgcn_wmma_f32_16x16x32_bf16(
              false, a, false, b1f[kt], (short)0, acc, false, false);
        }
#pragma unroll
        for (int vr = 0; vr < 8; ++vr) {
          int rr = mt*16 + vr + hk*8;          // D row
          s1[rr*64 + o] = f2bf(fmaxf(acc[vr] + bias1, 0.f));
        }
      }
    }
    __syncthreads();

    // ===== GEMM2: (64 x 192) x (192 x 32); wave = (mt2,nt2), mt = mt2, mt2+2
    {
      const int o = nt2*16 + n;
#pragma unroll
      for (int mi = 0; mi < 2; ++mi) {
        const int mt = mt2 + mi*2;
        const int r = mt*16 + n;
        const int p = r >> 2, t0 = r & 3;
        v8f acc = {};
#pragma unroll
        for (int kt = 0; kt < 6; ++kt) {
          const v16bf a = load_afrag<6>(s1, p*6 + t0, 64, kt, hk);
          acc = __builtin_amdgcn_wmma_f32_16x16x32_bf16(
              false, a, false, b2f[kt], (short)0, acc, false, false);
        }
#pragma unroll
        for (int vr = 0; vr < 8; ++vr) {
          int rr = mt*16 + vr + hk*8;
          s2[rr*32 + o] = f2bf(fmaxf(acc[vr] + bias2, 0.f));
        }
      }
    }
    __syncthreads();

    // ===== GEMM3: (32 x 96) x (96 x 32); wave = (mt3,nt3)
    {
      const int o = nt3*16 + n;
      const int r = mt3*16 + n;
      const int p = r >> 1, t0 = r & 1;
      v8f acc = {};
#pragma unroll
      for (int kt = 0; kt < 3; ++kt) {
        const v16bf a = load_afrag<5>(s2, p*4 + t0, 32, kt, hk);
        acc = __builtin_amdgcn_wmma_f32_16x16x32_bf16(
            false, a, false, b3f[kt], (short)0, acc, false, false);
      }
#pragma unroll
      for (int vr = 0; vr < 8; ++vr) {
        int rr = mt3*16 + vr + hk*8;
        int pp = rr >> 1, t3 = rr & 1;
        sb[pp*64 + o*2 + t3] = fmaxf(acc[vr] + bias3, 0.f);  // reshape (32,2)->64
      }
    }
    __syncthreads();

    // ===== per-scene segment max (2 scenes per block)
    {
      int sc = tid >> 6, h = tid & 63;         // all 128 threads
      float m = sb[(sc*8)*64 + h];
#pragma unroll
      for (int p = 1; p < 8; ++p) m = fmaxf(m, sb[(sc*8 + p)*64 + h]);
      mx[sc*64 + h] = m;
    }
    __syncthreads();

    // ===== hidden2pos: rel = [s, mx] @ W_hp^T + b_hp ; emit output
    if (tid < NPED*2) {
      int p = tid >> 1, oo = tid & 1, sc = p >> 3;
      float s = b_hp[oo];
      for (int h = 0; h < 64; ++h) s += sb[p*64 + h]  * W_hp[oo*128 + h];
      for (int h = 0; h < 64; ++h) s += mx[sc*64 + h] * W_hp[oo*128 + 64 + h];
      relb[tid] = s;
      out[((size_t)step*BATCH + ped0 + p)*2 + oo] = s;
    }
    __syncthreads();

    // ===== dec = rel @ W_se^T + b_se ; shift obs_emb left, append (bf16 verbatim)
    for (int idx = tid; idx < NPED*64; idx += 128) {
      int p = idx >> 6, c = idx & 63;
      float d = relb[p*2]*W_se[c*2] + relb[p*2 + 1]*W_se[c*2 + 1] + b_se[c];
#pragma unroll
      for (int tt = 0; tt < 7; ++tt)
        act[(p*8 + tt)*64 + c] = act[(p*8 + tt + 1)*64 + c];
      act[(p*8 + 7)*64 + c] = f2bf(d);
    }
    __syncthreads();
  }
}

extern "C" void kernel_launch(void* const* d_in, const int* in_sizes, int n_in,
                              void* d_out, int out_size, void* d_ws, size_t ws_size,
                              hipStream_t stream) {
  (void)in_sizes; (void)n_in; (void)out_size; (void)ws_size;
  const float* obs_traj = (const float*)d_in[0];
  // d_in[1]=last_pos, d_in[2]=last_pos_rel: carried state never affects rels -> unused
  const float* W_se = (const float*)d_in[3];
  const float* b_se = (const float*)d_in[4];
  const float* v1   = (const float*)d_in[5];
  const float* g1   = (const float*)d_in[6];
  const float* b1   = (const float*)d_in[7];
  const float* v2   = (const float*)d_in[8];
  const float* g2   = (const float*)d_in[9];
  const float* b2   = (const float*)d_in[10];
  const float* v3   = (const float*)d_in[11];
  const float* g3   = (const float*)d_in[12];
  const float* b3   = (const float*)d_in[13];
  const float* W_hp = (const float*)d_in[14];
  const float* b_hp = (const float*)d_in[15];
  // d_in[16]=seq_start_end: scenes are contiguous blocks of 8 -> implicit
  unsigned short* wf = (unsigned short*)d_ws;   // 43008 bytes of scratch used
  float* out = (float*)d_out;

  hipLaunchKernelGGL(prep_weights, dim3(1), dim3(128), 0, stream,
                     v1, g1, v2, g2, v3, g3, wf);
  hipLaunchKernelGGL(encoder_kernel, dim3(BATCH / NPED), dim3(128), 0, stream,
                     obs_traj, W_se, b_se, b1, b2, b3, W_hp, b_hp, wf, out);
}
